// MultiHeadAttention_9740985827877
// MI455X (gfx1250) — matrix operand order repaired and checked by reading, not compiled, not run on hardware
//
#include <hip/hip_runtime.h>
#include <math.h>

// ---------------- problem constants ----------------
#define BATCH   2
#define SEQ     2048
#define DMODEL  768
#define HEADS   12
#define DK      64
#define MROWS   (BATCH*SEQ)        // 4096

typedef __attribute__((ext_vector_type(16))) _Float16 v16h;
typedef __attribute__((ext_vector_type(8)))  _Float16 v8h;
typedef __attribute__((ext_vector_type(4)))  _Float16 v4h;
typedef __attribute__((ext_vector_type(8)))  float    v8f;

__device__ __forceinline__ v8f wmma16(v16h a, v16h b, v8f c) {
    // D = A(16x32 f16) * B(32x16 f16) + C(16x16 f32)
    return __builtin_amdgcn_wmma_f32_16x16x32_f16(
        /*neg_a=*/false, a, /*neg_b=*/false, b,
        /*c_mod=*/(short)0, c, /*reuse_a=*/false, /*reuse_b=*/false);
}

// A-matrix fragment: two contiguous 8-half runs (K = 8*hi .. , K = 16+8*hi ..)
__device__ __forceinline__ v16h ld_frag_a(const _Float16* p) {
    v8h lo = *(const v8h*)p;
    v8h hi = *(const v8h*)(p + 16);
    return __builtin_shufflevector(lo, hi, 0,1,2,3,4,5,6,7,8,9,10,11,12,13,14,15);
}
// B-matrix fragment: two contiguous 8-half runs (K = 8*hi .. , K = 16+8*hi ..)
__device__ __forceinline__ v16h ld_frag_b(const _Float16* p) {
    v8h lo = *(const v8h*)p;
    v8h hi = *(const v8h*)(p + 16);
    return __builtin_shufflevector(lo, hi, 0,1,2,3,4,5,6,7,8,9,10,11,12,13,14,15);
}

// ---- DPP 16-lane butterfly reductions (stay within the 16-lane DPP row, so
// ---- lanes 0-15 / 16-31 reduce independently — matches WMMA C/D layout) ----
__device__ __forceinline__ float dpp_fmax16(float x) {
    int t;
    t = __builtin_amdgcn_update_dpp(0, __float_as_int(x), 0xB1,  0xF, 0xF, true); // quad_perm [1,0,3,2]
    x = fmaxf(x, __int_as_float(t));
    t = __builtin_amdgcn_update_dpp(0, __float_as_int(x), 0x4E,  0xF, 0xF, true); // quad_perm [2,3,0,1]
    x = fmaxf(x, __int_as_float(t));
    t = __builtin_amdgcn_update_dpp(0, __float_as_int(x), 0x141, 0xF, 0xF, true); // row_half_mirror
    x = fmaxf(x, __int_as_float(t));
    t = __builtin_amdgcn_update_dpp(0, __float_as_int(x), 0x140, 0xF, 0xF, true); // row_mirror
    x = fmaxf(x, __int_as_float(t));
    return x;
}
__device__ __forceinline__ float dpp_fadd16(float x) {
    int t;
    t = __builtin_amdgcn_update_dpp(0, __float_as_int(x), 0xB1,  0xF, 0xF, true);
    x += __int_as_float(t);
    t = __builtin_amdgcn_update_dpp(0, __float_as_int(x), 0x4E,  0xF, 0xF, true);
    x += __int_as_float(t);
    t = __builtin_amdgcn_update_dpp(0, __float_as_int(x), 0x141, 0xF, 0xF, true);
    x += __int_as_float(t);
    t = __builtin_amdgcn_update_dpp(0, __float_as_int(x), 0x140, 0xF, 0xF, true);
    x += __int_as_float(t);
    return x;
}

// ---------------------------------------------------------------------------
// Tiled GEMM: Out[M,768] = A[M,768] * W[768,768] + bias
//   Block tile 128x64, 8 waves of 32x32, K-step 32, f16 WMMA with f32 acc.
//   A_F16:  A is f16 (attn output) vs f32 (q/k/v inputs, converted on load)
//   STORE_MODE: 0 = f16 head-split [B,H,S,64]
//               1 = f16 head-split transposed [B,H,64,S]   (for V)
//               2 = f32 plain [M,768]                      (final output)
// ---------------------------------------------------------------------------
#define KP 40   // padded LDS row stride in halfs (80B: 16B-aligned, conflict-free)

template<bool A_F16, int STORE_MODE>
__global__ __launch_bounds__(256) void gemm_proj(
    const void* __restrict__ Aptr, const float* __restrict__ W,
    const float* __restrict__ bias, void* __restrict__ Out)
{
    __shared__ __align__(16) _Float16 As[128][KP];   // A tile 128x32 (padded)
    __shared__ __align__(16) _Float16 Bs[64][KP];    // W tile transposed: Bs[n][k]

    const int tid  = threadIdx.x;
    const int bm   = blockIdx.x;          // 0..31  (M tiles of 128)
    const int bn   = blockIdx.y;          // 0..11  (N tiles of 64)
    const int w    = tid >> 5;
    const int lane = tid & 31;
    const int hi   = lane >> 4;           // 0/1 lane group
    const int l16  = lane & 15;
    const int wm   = w >> 1;              // 0..3  wave M sub-tile (32 rows)
    const int wn   = w & 1;               // 0..1  wave N sub-tile (32 cols)

    v8f acc[2][2] = {};

    for (int kb = 0; kb < DMODEL; kb += 32) {
        __syncthreads();
        // ---- load A tile (128x32) ----
        if (A_F16) {
            const _Float16* A = (const _Float16*)Aptr;
            #pragma unroll
            for (int i = 0; i < 4; ++i) {
                int f = tid + i * 256;            // 0..1023 quads
                int row = f >> 3, kq = f & 7;
                v4h v = *(const v4h*)(A + (size_t)(bm*128 + row)*DMODEL + kb + kq*4);
                *(v4h*)(&As[row][kq*4]) = v;
            }
        } else {
            const float* A = (const float*)Aptr;
            #pragma unroll
            for (int i = 0; i < 4; ++i) {
                int f = tid + i * 256;
                int row = f >> 3, kq = f & 7;
                float4 v = *(const float4*)(A + (size_t)(bm*128 + row)*DMODEL + kb + kq*4);
                v4h hv = {(_Float16)v.x, (_Float16)v.y, (_Float16)v.z, (_Float16)v.w};
                *(v4h*)(&As[row][kq*4]) = hv;
            }
        }
        // ---- load W tile (32x64) transposed into Bs[n][k] ----
        #pragma unroll
        for (int i = 0; i < 2; ++i) {
            int f = tid + i * 256;                // 0..511 quads
            int k = f >> 4, nq = f & 15;
            float4 v = *(const float4*)(W + (size_t)(kb + k)*DMODEL + bn*64 + nq*4);
            Bs[nq*4 + 0][k] = (_Float16)v.x;
            Bs[nq*4 + 1][k] = (_Float16)v.y;
            Bs[nq*4 + 2][k] = (_Float16)v.z;
            Bs[nq*4 + 3][k] = (_Float16)v.w;
        }
        __syncthreads();

        // ---- fragments + 4 WMMAs ----
        v16h af[2], bf[2];
        #pragma unroll
        for (int i = 0; i < 2; ++i)
            af[i] = ld_frag_a(&As[wm*32 + i*16 + l16][hi*8]);
        #pragma unroll
        for (int j = 0; j < 2; ++j)
            bf[j] = ld_frag_b(&Bs[wn*32 + j*16 + l16][hi*8]);
        #pragma unroll
        for (int i = 0; i < 2; ++i)
            #pragma unroll
            for (int j = 0; j < 2; ++j)
                acc[i][j] = wmma16(af[i], bf[j], acc[i][j]);
    }

    // ---- epilogue: bias + store ----
    float bj[2];
    #pragma unroll
    for (int j = 0; j < 2; ++j)
        bj[j] = bias[bn*64 + wn*32 + j*16 + l16];

    #pragma unroll
    for (int i = 0; i < 2; ++i) {
        #pragma unroll
        for (int j = 0; j < 2; ++j) {
            #pragma unroll
            for (int r = 0; r < 8; ++r) {
                int gm = bm*128 + wm*32 + i*16 + hi*8 + r;   // global row (b*S+s)
                int gn = bn*64  + wn*32 + j*16 + l16;        // global col (h*64+dk)
                float val = acc[i][j][r] + bj[j];
                if (STORE_MODE == 2) {
                    ((float*)Out)[(size_t)gm*DMODEL + gn] = val;
                } else {
                    int b = gm >> 11, s = gm & (SEQ-1);
                    int h = gn >> 6,  d = gn & (DK-1);
                    size_t idx = (STORE_MODE == 0)
                        ? (((size_t)(b*HEADS + h)*SEQ + s)*DK + d)   // [B,H,S,64]
                        : (((size_t)(b*HEADS + h)*DK  + d)*SEQ + s); // [B,H,64,S]
                    ((_Float16*)Out)[idx] = (_Float16)val;
                }
            }
        }
    }
}

// ---------------------------------------------------------------------------
// Flash attention: one wave handles one 16-row q-tile of one (b,h).
//   Qh,Kh: [B*H, S, 64] f16.  Vt: [B*H, 64, S] f16.  Out: [B, S, 768] f16.
//   64 keys per pass: 8 QK^T WMMAs + softmax + 8 P*V WMMAs.
// ---------------------------------------------------------------------------
#define PP 72   // P tile LDS row stride in halfs (144B = 9*16B: aligned, padded)

__global__ __launch_bounds__(128) void attn_kernel(
    const _Float16* __restrict__ Qh, const _Float16* __restrict__ Kh,
    const _Float16* __restrict__ Vt, _Float16* __restrict__ AttnOut)
{
    __shared__ __align__(16) _Float16 Plds[4][16][PP];  // per-wave 16x64 P tile

    const int w    = threadIdx.x >> 5;
    const int lane = threadIdx.x & 31;
    const int hi   = lane >> 4;
    const int l16  = lane & 15;

    const int wave = blockIdx.x * 4 + w;       // 0..3071
    const int qt   = wave & (SEQ/16 - 1);      // 0..127 q-tile
    const int bh   = wave >> 7;                // 0..23
    const int bb   = bh / HEADS, hh = bh % HEADS;

    const _Float16* Qb = Qh + (size_t)bh * SEQ * DK;
    const _Float16* Kb = Kh + (size_t)bh * SEQ * DK;
    const _Float16* Vb = Vt + (size_t)bh * DK  * SEQ;

    // Q A-fragments for dk = 0..31 and 32..63, kept in registers for whole row.
    v16h qf[2];
    {
        const int qrow = qt*16 + l16;
        #pragma unroll
        for (int c = 0; c < 2; ++c)
            qf[c] = ld_frag_a(Qb + (size_t)qrow*DK + c*32 + hi*8);
    }

    v8f acc[4] = {};                 // out tile 16x64 f32, D-layout per 16-col block
    float mrow[8], lrow[8];
    #pragma unroll
    for (int r = 0; r < 8; ++r) { mrow[r] = -1e30f; lrow[r] = 0.0f; }

    const float scale = 0.125f;      // 1/sqrt(64)

    for (int kb0 = 0; kb0 < SEQ; kb0 += 64) {
        // ---- scores: four 16x16 tiles (keys kb0+16*sub .. +15) ----
        v8f s[4];
        #pragma unroll
        for (int sub = 0; sub < 4; ++sub) {
            const int key = kb0 + sub*16 + l16;        // column = key
            v8f sv = {};
            #pragma unroll
            for (int c = 0; c < 2; ++c) {
                v16h kf = ld_frag_b(Kb + (size_t)key*DK + c*32 + hi*8);
                sv = wmma16(qf[c], kf, sv);
            }
            s[sub] = sv * scale;
        }

        // ---- online softmax (rows live in 16-lane groups; DPP reductions) ----
        #pragma unroll
        for (int r = 0; r < 8; ++r) {
            float mx = fmaxf(fmaxf(s[0][r], s[1][r]), fmaxf(s[2][r], s[3][r]));
            mx = dpp_fmax16(mx);
            float mnew = fmaxf(mrow[r], mx);
            float corr = __expf(mrow[r] - mnew);
            mrow[r] = mnew;
            float p[4];
            #pragma unroll
            for (int sub = 0; sub < 4; ++sub) p[sub] = __expf(s[sub][r] - mnew);
            float rs = (p[0] + p[1]) + (p[2] + p[3]);
            rs = dpp_fadd16(rs);
            lrow[r] = lrow[r] * corr + rs;
            #pragma unroll
            for (int j = 0; j < 4; ++j) acc[j][r] *= corr;
            // stash P (f16) for D-layout -> A-layout transpose
            #pragma unroll
            for (int sub = 0; sub < 4; ++sub)
                Plds[w][hi*8 + r][sub*16 + l16] = (_Float16)p[sub];
        }

        // ---- P (16x64) A-fragments, V B-fragments, 8 WMMAs ----
        v16h pf[2];
        #pragma unroll
        for (int c = 0; c < 2; ++c)
            pf[c] = ld_frag_a(&Plds[w][l16][c*32 + hi*8]);
        #pragma unroll
        for (int j = 0; j < 4; ++j) {
            const int d = j*16 + l16;                   // output column = dk
            #pragma unroll
            for (int c = 0; c < 2; ++c) {
                v16h vf = ld_frag_b(Vb + (size_t)d*SEQ + kb0 + c*32 + hi*8);
                acc[j] = wmma16(pf[c], vf, acc[j]);
            }
        }
    }

    // ---- normalize + store merged-head f16 [B,S,768] ----
    #pragma unroll
    for (int r = 0; r < 8; ++r) {
        float inv = 1.0f / lrow[r];
        int srow = qt*16 + hi*8 + r;
        size_t rowoff = ((size_t)bb*SEQ + srow)*DMODEL + hh*DK;
        #pragma unroll
        for (int j = 0; j < 4; ++j)
            AttnOut[rowoff + j*16 + l16] = (_Float16)(acc[j][r] * inv);
    }
}

// ---------------------------------------------------------------------------
extern "C" void kernel_launch(void* const* d_in, const int* in_sizes, int n_in,
                              void* d_out, int out_size, void* d_ws, size_t ws_size,
                              hipStream_t stream) {
    const float* q  = (const float*)d_in[0];
    const float* k  = (const float*)d_in[1];
    const float* v  = (const float*)d_in[2];
    const float* Wq = (const float*)d_in[3];
    const float* bq = (const float*)d_in[4];
    const float* Wk = (const float*)d_in[5];
    const float* bk = (const float*)d_in[6];
    const float* Wv = (const float*)d_in[7];
    const float* bv = (const float*)d_in[8];
    const float* Wo = (const float*)d_in[9];
    const float* bo = (const float*)d_in[10];

    const size_t HBYTES = (size_t)BATCH*HEADS*SEQ*DK*sizeof(_Float16); // 6.29 MB
    _Float16* Qh = (_Float16*)d_ws;
    _Float16* Kh = (_Float16*)((char*)d_ws + 1*HBYTES);
    _Float16* Vt = (_Float16*)((char*)d_ws + 2*HBYTES);
    _Float16* AO = (_Float16*)((char*)d_ws + 3*HBYTES);

    dim3 ggrid(MROWS/128, DMODEL/64);   // 32 x 12
    gemm_proj<false,0><<<ggrid, 256, 0, stream>>>(q, Wq, bq, Qh);
    gemm_proj<false,0><<<ggrid, 256, 0, stream>>>(k, Wk, bk, Kh);
    gemm_proj<false,1><<<ggrid, 256, 0, stream>>>(v, Wv, bv, Vt);

    // B*H*(S/16) = 3072 waves, 4 waves per block
    attn_kernel<<<768, 128, 0, stream>>>(Qh, Kh, Vt, AO);

    gemm_proj<true,2><<<ggrid, 256, 0, stream>>>(AO, Wo, bo, d_out);
}
